// PagedHSTUInferLayer_79250736545886
// MI455X (gfx1250) — matrix operand order repaired and checked by reading, not compiled, not run on hardware
//
#include <hip/hip_runtime.h>
#include <math.h>

#define NH   4
#define DH   128
#define D_   512
#define U_   2048
#define B_   32
#define S_   256
#define H_   2048
#define T_   (H_ + S_)   /* 2304 */
#define NT   8192
#define EPS  1e-5f
#define ALPHA 0.08838834764831845f  /* 1/sqrt(128) */

typedef __attribute__((ext_vector_type(16))) __bf16 v16bf;
typedef __attribute__((ext_vector_type(4)))  __bf16 v4bf;
typedef __attribute__((ext_vector_type(8)))  float  v8f;
typedef __attribute__((ext_vector_type(4)))  unsigned int u32x4;
typedef __attribute__((ext_vector_type(4)))  int v4i;

#define DEV static __device__ __forceinline__

// ---- CDNA5 async global->LDS copy (ASYNCcnt path), with sync fallback ----
#if __has_builtin(__builtin_amdgcn_global_load_async_to_lds_b128)
#define HAS_ASYNC 1
#define ASYNC_CP16(g, l)                                                   \
  __builtin_amdgcn_global_load_async_to_lds_b128(                          \
      (__attribute__((address_space(1))) v4i*)(g),                         \
      (__attribute__((address_space(3))) v4i*)(l), 0, 0)
#else
#define HAS_ASYNC 0
#define ASYNC_CP16(g, l) (*(u32x4*)(l) = *(const u32x4*)(g))
#endif

DEV void wait_async8() {
#if HAS_ASYNC
#if __has_builtin(__builtin_amdgcn_s_wait_asynccnt)
  __builtin_amdgcn_s_wait_asynccnt(8);
#else
  asm volatile("s_wait_asynccnt 0x8" ::: "memory");
#endif
#endif
}

// ---- WMMA operand loaders (CDNA5 16x16x32 bf16 layouts, wave32) ----
// A tile 16x32 (MxK), source row-major, leading dim `ld`:
//   lane<16 : row M=lane,    elems K = {0..7, 16..23}
//   lane>=16: row M=lane-16, elems K = {8..15, 24..31}
DEV v16bf ld_a(const __bf16* base, int ld, int lane) {
  const __bf16* p = base + (lane & 15) * ld + ((lane >> 4) << 3);
  v16bf r;
  *reinterpret_cast<u32x4*>(&r)       = *reinterpret_cast<const u32x4*>(p);
  *(reinterpret_cast<u32x4*>(&r) + 1) = *reinterpret_cast<const u32x4*>(p + 16);
  return r;
}
// B tile 32x16 (KxN), source stored N-major (src[n][k], leading dim `ld`):
//   lane<16 : col N=lane,    rows K = {0..7, 16..23}
//   lane>=16: col N=lane-16, rows K = {8..15, 24..31}
DEV v16bf ld_b(const __bf16* base, int ld, int lane) {
  const __bf16* p = base + (lane & 15) * ld + ((lane >> 4) << 3);
  v16bf r;
  *reinterpret_cast<u32x4*>(&r)       = *reinterpret_cast<const u32x4*>(p);
  *(reinterpret_cast<u32x4*>(&r) + 1) = *reinterpret_cast<const u32x4*>(p + 16);
  return r;
}
DEV v8f wmma_bf16(v16bf a, v16bf b, v8f c) {
  return __builtin_amdgcn_wmma_f32_16x16x32_bf16(false, a, false, b, (short)0, c,
                                                 false, false);
}
DEV float silu(float x) { return x / (1.f + __expf(-x)); }

// ---- 1. LayerNorm(input) -> bf16, one wave per row of 512 ----
__global__ __launch_bounds__(256) void k_ln_in(const float* __restrict__ x,
    const float* __restrict__ w, const float* __restrict__ bb,
    __bf16* __restrict__ out)
{
  int wave = (blockIdx.x * 256 + threadIdx.x) >> 5;
  int lane = threadIdx.x & 31;
  const float* row = x + (size_t)wave * D_;
  float4 v4[4];
  const float4* rp = (const float4*)(row + lane * 16);
#pragma unroll
  for (int i = 0; i < 4; ++i) v4[i] = rp[i];
  const float* vv = (const float*)v4;
  float s = 0.f, ss = 0.f;
#pragma unroll
  for (int i = 0; i < 16; ++i) { s += vv[i]; ss += vv[i] * vv[i]; }
#pragma unroll
  for (int m = 16; m; m >>= 1) { s += __shfl_xor(s, m); ss += __shfl_xor(ss, m); }
  float mean = s * (1.f / D_);
  float rs = rsqrtf(ss * (1.f / D_) - mean * mean + EPS);
  int base = lane * 16;
#pragma unroll
  for (int i = 0; i < 16; ++i)
    out[(size_t)wave * D_ + base + i] =
        (__bf16)((vv[i] - mean) * rs * w[base + i] + bb[base + i]);
}

// ---- 2. fp32 [R][C] -> bf16 [C][R] transpose (weights -> N-major) ----
__global__ __launch_bounds__(256) void k_tr(const float* __restrict__ in,
    __bf16* __restrict__ out, int R, int C)
{
  long idx = (long)blockIdx.x * 256 + threadIdx.x;
  if (idx >= (long)R * C) return;
  int c = (int)(idx / R), r = (int)(idx % R);
  out[idx] = (__bf16)in[(long)r * C + c];
}

// ---- 3. history fp32 [B,H,NH,DH] -> K bf16 [B,NH,T,DH] and V^T bf16 [B,NH,DH,T]
__global__ __launch_bounds__(256) void k_hist(const float* __restrict__ hk,
    const float* __restrict__ hv, __bf16* __restrict__ Kf, __bf16* __restrict__ Vt)
{
  long idx4 = ((long)blockIdx.x * 256 + threadIdx.x) * 4;  // over B*NH*H*DH
  int d = (int)(idx4 & 127);
  int t = (int)((idx4 >> 7) & (H_ - 1));
  int h = (int)((idx4 >> 18) & 3);
  int b = (int)(idx4 >> 20);
  long src = (((long)b * H_ + t) * NH + h) * DH + d;
  float4 kk = *(const float4*)(hk + src);
  float4 vv = *(const float4*)(hv + src);
  v4bf kp = { (__bf16)kk.x, (__bf16)kk.y, (__bf16)kk.z, (__bf16)kk.w };
  *(v4bf*)(Kf + ((long)(b * NH + h) * T_ + t) * DH + d) = kp;
  __bf16* vt = Vt + ((long)(b * NH + h) * DH + d) * T_ + t;
  vt[0 * T_] = (__bf16)vv.x;
  vt[1 * T_] = (__bf16)vv.y;
  vt[2 * T_] = (__bf16)vv.z;
  vt[3 * T_] = (__bf16)vv.w;
}

// ---- shared pipelined GEMM core: 16xK @ Wt -> 16x64 tile in acc[4] ----
DEV void gemm_16x64(const __bf16* __restrict__ Arow, const __bf16* __restrict__ Wt,
                    int n64, int lane, v8f acc[4])
{
  v16bf a_cur = ld_a(Arow, D_, lane);
  v16bf b_cur[4];
#pragma unroll
  for (int j = 0; j < 4; ++j)
    b_cur[j] = ld_b(Wt + (size_t)(n64 + j * 16) * D_, D_, lane);
#pragma unroll
  for (int k0 = 0; k0 < D_; k0 += 32) {
    int kn = (k0 + 32 < D_) ? k0 + 32 : k0;       // clamp on last iter
    v16bf a_nxt = ld_a(Arow + kn, D_, lane);
    v16bf b_nxt[4];
#pragma unroll
    for (int j = 0; j < 4; ++j)
      b_nxt[j] = ld_b(Wt + (size_t)(n64 + j * 16) * D_ + kn, D_, lane);
#pragma unroll
    for (int j = 0; j < 4; ++j) acc[j] = wmma_bf16(a_cur, b_cur[j], acc[j]);
    a_cur = a_nxt;
#pragma unroll
    for (int j = 0; j < 4; ++j) b_cur[j] = b_nxt[j];
  }
}

// ---- 4. uvqk GEMM: [NT,512]bf16 @ Wt[2048,512]bf16 -> bias+silu -> scatter ----
__global__ __launch_bounds__(256) void k_gemm1(const __bf16* __restrict__ X,
    const __bf16* __restrict__ Wt, const float* __restrict__ bias,
    __bf16* __restrict__ u_bf, __bf16* __restrict__ q_ws,
    __bf16* __restrict__ Kf, __bf16* __restrict__ Vt)
{
  int lane = threadIdx.x & 31;
  int wid  = (blockIdx.x * 256 + threadIdx.x) >> 5;
  int m    = wid >> 5;             // 512 M-tiles
  int n64  = (wid & 31) << 6;      // 32 N-tiles of 64
  v8f acc[4] = {};
  gemm_16x64(X + (size_t)m * 16 * D_, Wt, n64, lane, acc);
#pragma unroll
  for (int j = 0; j < 4; ++j) {
#pragma unroll
    for (int r = 0; r < 8; ++r) {
      int row = m * 16 + r + ((lane >> 4) << 3);
      int col = n64 + j * 16 + (lane & 15);
      float sv = silu(acc[j][r] + bias[col]);
      int bi = row >> 8, s = row & 255;
      if (col < 512) {                                   // u
        u_bf[(size_t)row * D_ + col] = (__bf16)sv;
      } else if (col < 1024) {                           // v -> V^T
        int cv = col - 512, h = cv >> 7, d = cv & 127;
        Vt[((size_t)(bi * NH + h) * DH + d) * T_ + (H_ + s)] = (__bf16)sv;
      } else if (col < 1536) {                           // q
        int cv = col - 1024, h = cv >> 7, d = cv & 127;
        q_ws[((size_t)(bi * NH + h) * S_ + s) * DH + d] = (__bf16)sv;
      } else {                                           // k appended after history
        int cv = col - 1536, h = cv >> 7, d = cv & 127;
        Kf[((size_t)(bi * NH + h) * T_ + (H_ + s)) * DH + d] = (__bf16)sv;
      }
    }
  }
}

// ---- 5. HSTU attention, LDS-staged & async double-buffered ----
// Block = 4 waves = 64 queries of one (b,h). Each 32-key chunk of K and V^T is
// cooperatively copied into LDS (async), shared by all 4 waves.
#define KLD 136                       /* 128 + 8 pad: kills bank conflicts  */
#define VLD 40                        /* 32 + 8 pad                         */

DEV void stage_chunk(const __bf16* __restrict__ Kb, const __bf16* __restrict__ Vb,
                     int t0, __bf16* kl, __bf16* vl, int tid)
{
#pragma unroll
  for (int i = 0; i < 4; ++i) {       // K chunk: 32 rows x 256B = 512 x 16B
    int id = i * 128 + tid;
    int row = id >> 4, c = id & 15;
    ASYNC_CP16(Kb + (size_t)(t0 + row) * DH + c * 8, kl + row * KLD + c * 8);
  }
#pragma unroll
  for (int i = 0; i < 4; ++i) {       // V^T chunk: 128 rows x 64B = 512 x 16B
    int id = i * 128 + tid;
    int row = id >> 2, c = id & 3;
    ASYNC_CP16(Vb + (size_t)row * T_ + t0 + c * 8, vl + row * VLD + c * 8);
  }
}

__global__ __launch_bounds__(128) void k_attn(const __bf16* __restrict__ q_ws,
    const __bf16* __restrict__ Kf, const __bf16* __restrict__ Vt,
    float* __restrict__ attn)
{
  __shared__ alignas(16) __bf16 Klds[2][32 * KLD];
  __shared__ alignas(16) __bf16 Vlds[2][128 * VLD];
  __shared__ alignas(16) __bf16 Plds[4][16 * 40];

  int tid  = threadIdx.x;
  int lane = tid & 31;
  int widx = tid >> 5;
  int blk = blockIdx.x;               // over B*NH*(S/64) = 512
  int b  = blk >> 4;
  int h  = (blk >> 2) & 3;
  int sq = (blk & 3) << 6;            // 64-query group base
  int s0 = sq + widx * 16;            // this wave's query tile

  const __bf16* qb = q_ws + ((size_t)(b * NH + h) * S_ + s0) * DH;
  const __bf16* Kb = Kf + (size_t)(b * NH + h) * T_ * DH;
  const __bf16* Vb = Vt + (size_t)(b * NH + h) * DH * T_;
  __bf16* pl = Plds[widx];

  v16bf aq[4];
#pragma unroll
  for (int c = 0; c < 4; ++c) aq[c] = ld_a(qb + c * 32, DH, lane);
  v8f o[8] = {};

  int tmax = H_ + sq + 64;            // covers the block's highest query pos
  if (tmax > T_) tmax = T_;           // already a multiple of 32

  stage_chunk(Kb, Vb, 0, Klds[0], Vlds[0], tid);
  for (int t0 = 0, ci = 0; t0 < tmax; t0 += 32, ++ci) {
    int cur = ci & 1;
    int tn = (t0 + 32 < tmax) ? t0 + 32 : t0;   // clamp; keeps wait count exact
    stage_chunk(Kb, Vb, tn, Klds[cur ^ 1], Vlds[cur ^ 1], tid);
    wait_async8();                    // chunk ci resident; ci+1 still in flight
    __syncthreads();

    const __bf16* kl = Klds[cur];
    const __bf16* vl = Vlds[cur];
    // scores: K-dim = dh(128); two 16-key subtiles per chunk
    v8f sc0 = {}, sc1 = {};
#pragma unroll
    for (int c = 0; c < 4; ++c) {
      sc0 = wmma_bf16(aq[c], ld_b(kl + c * 32, KLD, lane), sc0);
      sc1 = wmma_bf16(aq[c], ld_b(kl + 16 * KLD + c * 32, KLD, lane), sc1);
    }
    // mask (key pos <= H + query idx) + silu(alpha*s); P -> LDS -> A-operand
#pragma unroll
    for (int r = 0; r < 8; ++r) {
      int qpos = H_ + s0 + r + ((lane >> 4) << 3);
      int row  = r + ((lane >> 4) << 3);
      float x0 = sc0[r] * ALPHA;
      float x1 = sc1[r] * ALPHA;
      float p0 = (t0 + (lane & 15) <= qpos) ? silu(x0) : 0.f;
      float p1 = (t0 + 16 + (lane & 15) <= qpos) ? silu(x1) : 0.f;
      pl[row * 40 + (lane & 15)]      = (__bf16)p0;
      pl[row * 40 + 16 + (lane & 15)] = (__bf16)p1;
    }
    v16bf ap = ld_a(pl, 40, lane);    // 16x32 P, K-dim = keys
#pragma unroll
    for (int j = 0; j < 8; ++j)
      o[j] = wmma_bf16(ap, ld_b(vl + (j * 16) * VLD, VLD, lane), o[j]);
    __syncthreads();                  // all waves done before buffer reuse
  }
#pragma unroll
  for (int j = 0; j < 8; ++j)
#pragma unroll
    for (int r = 0; r < 8; ++r) {
      int row = b * S_ + s0 + r + ((lane >> 4) << 3);
      attn[(size_t)row * D_ + h * DH + j * 16 + (lane & 15)] = o[j][r];
    }
}

// ---- 6. gated output LN: par = u * LN(attn) -> bf16 ----
__global__ __launch_bounds__(256) void k_ln_out(const float* __restrict__ x,
    const float* __restrict__ w, const float* __restrict__ bb,
    const __bf16* __restrict__ u_bf, __bf16* __restrict__ out)
{
  int wave = (blockIdx.x * 256 + threadIdx.x) >> 5;
  int lane = threadIdx.x & 31;
  const float* row = x + (size_t)wave * D_;
  float4 v4[4];
  const float4* rp = (const float4*)(row + lane * 16);
#pragma unroll
  for (int i = 0; i < 4; ++i) v4[i] = rp[i];
  const float* vv = (const float*)v4;
  float s = 0.f, ss = 0.f;
#pragma unroll
  for (int i = 0; i < 16; ++i) { s += vv[i]; ss += vv[i] * vv[i]; }
#pragma unroll
  for (int m = 16; m; m >>= 1) { s += __shfl_xor(s, m); ss += __shfl_xor(ss, m); }
  float mean = s * (1.f / D_);
  float rs = rsqrtf(ss * (1.f / D_) - mean * mean + EPS);
  int base = lane * 16;
#pragma unroll
  for (int i = 0; i < 16; ++i) {
    size_t idx = (size_t)wave * D_ + base + i;
    float ln = (vv[i] - mean) * rs * w[base + i] + bb[base + i];
    out[idx] = (__bf16)((float)u_bf[idx] * ln);
  }
}

// ---- 7. out projection + residual: [NT,512]bf16 @ Wt2[512,512]bf16 + input ----
__global__ __launch_bounds__(256) void k_gemm2(const __bf16* __restrict__ P,
    const __bf16* __restrict__ Wt, const float* __restrict__ resid,
    float* __restrict__ out)
{
  int lane = threadIdx.x & 31;
  int wid  = (blockIdx.x * 256 + threadIdx.x) >> 5;
  int m    = wid >> 3;
  int n64  = (wid & 7) << 6;
  v8f acc[4] = {};
  gemm_16x64(P + (size_t)m * 16 * D_, Wt, n64, lane, acc);
#pragma unroll
  for (int j = 0; j < 4; ++j)
#pragma unroll
    for (int r = 0; r < 8; ++r) {
      int row = m * 16 + r + ((lane >> 4) << 3);
      int col = n64 + j * 16 + (lane & 15);
      size_t idx = (size_t)row * D_ + col;
      out[idx] = acc[j][r] + resid[idx];
    }
}

extern "C" void kernel_launch(void* const* d_in, const int* in_sizes, int n_in,
                              void* d_out, int out_size, void* d_ws, size_t ws_size,
                              hipStream_t stream)
{
  (void)in_sizes; (void)n_in; (void)out_size; (void)ws_size;
  const float* layer_input = (const float*)d_in[0];
  const float* hist_k      = (const float*)d_in[1];
  const float* hist_v      = (const float*)d_in[2];
  const float* ln_in_w     = (const float*)d_in[3];
  const float* ln_in_b     = (const float*)d_in[4];
  const float* W_uvqk      = (const float*)d_in[5];
  const float* b_uvqk      = (const float*)d_in[6];
  const float* ln_out_w    = (const float*)d_in[7];
  const float* ln_out_b    = (const float*)d_in[8];
  const float* W_proj      = (const float*)d_in[9];
  float* out = (float*)d_out;

  char* ws = (char*)d_ws;
  size_t off = 0;
  auto take = [&](size_t bytes) -> char* {
    char* p = ws + off; off += (bytes + 255) & ~(size_t)255; return p;
  };
  __bf16* x_bf = (__bf16*)take((size_t)NT * D_ * 2);
  __bf16* Wt1  = (__bf16*)take((size_t)U_ * D_ * 2);
  __bf16* Wt2  = (__bf16*)take((size_t)D_ * D_ * 2);
  __bf16* q_ws = (__bf16*)take((size_t)B_ * NH * S_ * DH * 2);
  __bf16* Kf   = (__bf16*)take((size_t)B_ * NH * T_ * DH * 2);
  __bf16* Vt   = (__bf16*)take((size_t)B_ * NH * T_ * DH * 2);
  __bf16* u_bf = (__bf16*)take((size_t)NT * D_ * 2);
  float*  attn = (float*) take((size_t)NT * D_ * 4);
  __bf16* par  = (__bf16*)take((size_t)NT * D_ * 2);

  k_ln_in<<<NT / 8, 256, 0, stream>>>(layer_input, ln_in_w, ln_in_b, x_bf);
  k_tr<<<(D_ * U_) / 256, 256, 0, stream>>>(W_uvqk, Wt1, D_, U_);
  k_tr<<<(D_ * D_) / 256, 256, 0, stream>>>(W_proj, Wt2, D_, D_);
  k_hist<<<(B_ * NH * H_ * DH) / 1024, 256, 0, stream>>>(hist_k, hist_v, Kf, Vt);
  k_gemm1<<<(NT / 16) * (U_ / 64) / 8, 256, 0, stream>>>(x_bf, Wt1, b_uvqk,
                                                         u_bf, q_ws, Kf, Vt);
  k_attn<<<B_ * NH * (S_ / 64), 128, 0, stream>>>(q_ws, Kf, Vt, attn);
  k_ln_out<<<NT / 8, 256, 0, stream>>>(attn, ln_out_w, ln_out_b, u_bf, par);
  k_gemm2<<<(NT / 16) * (D_ / 64) / 8, 256, 0, stream>>>(par, Wt2, layer_input, out);
}
